// SAGE_w_feat_43396349559018
// MI455X (gfx1250) — matrix operand order repaired and checked by reading, not compiled, not run on hardware
//
#include <hip/hip_runtime.h>
#include <math.h>

#define NN 100000
#define EE 1600000
#define LDA 136  // 128 halves + 8 pad -> bank spread, rows stay 16B aligned

typedef __attribute__((ext_vector_type(16))) _Float16 v16h;
typedef __attribute__((ext_vector_type(8)))  _Float16 v8h;
typedef __attribute__((ext_vector_type(8)))  float    v8f;

__device__ __forceinline__ v16h cat16(v8h lo, v8h hi) {
  v16h r;
#pragma unroll
  for (int i = 0; i < 8; ++i) { r[i] = lo[i]; r[i + 8] = hi[i]; }
  return r;
}

// ---------------------------------------------------------------------------
// degree: deg[d] = sum over edges (float)
__global__ void degree_kernel(const int* __restrict__ dst, float* __restrict__ deg) {
  int i = blockIdx.x * blockDim.x + threadIdx.x;
  if (i < EE) atomicAdd(&deg[dst[i]], 1.0f);
}

// ---------------------------------------------------------------------------
// scatter-add: agg[dst] += feat[src]; one edge per wave per iteration,
// float4 per lane. Feature matrices <= 51.2 MB -> L2-resident (192 MB L2).
// Grid-stride over 4 edges/wave with global_prefetch of the next source row
// to hide the random-gather latency behind the in-flight atomics.
#define EDGES_PER_WAVE 4
__global__ void scatter_kernel(const float* __restrict__ feat,
                               const int* __restrict__ src,
                               const int* __restrict__ dst,
                               float* __restrict__ agg, int C) {
  const int gid    = blockIdx.x * blockDim.x + threadIdx.x;
  const int w      = gid >> 5;
  const int lane   = gid & 31;
  const int stride = EE / EDGES_PER_WAVE;  // total waves launched
  const int chunks = C >> 2;

  for (int e = w; e < EE; e += stride) {
    int s = src[e], d = dst[e];
    int enx = e + stride;
    if (lane < chunks) {
      if (enx < EE) {
        // gfx1250 global_prefetch_b8: pull next edge's source row toward L2/L0
        __builtin_prefetch(feat + (size_t)src[enx] * C + lane * 4, 0, 3);
      }
      const float4 v = *(const float4*)(feat + (size_t)s * C + lane * 4);
      float* p = agg + (size_t)d * C + lane * 4;
      atomicAdd(p + 0, v.x);
      atomicAdd(p + 1, v.y);
      atomicAdd(p + 2, v.z);
      atomicAdd(p + 3, v.w);
    }
  }
}

// ---------------------------------------------------------------------------
// Fused SAGE GEMM: out[m, n] = (A1[m]/max(deg,1)) . W1[n] + A2[m] . W2[n] + bias[n]
// One 16-row node tile per block; one 16x16 output tile per wave via
// v_wmma_f32_16x16x32_f16 (f16 in, f32 accumulate). CIN/HAS1 are template
// params so the k-loop is a straight-line, branch-free chain of WMMAs.
template <int CIN, bool HAS1>
__global__ void __launch_bounds__(128)
sage_gemm(const float* __restrict__ A1, const float* __restrict__ deg,
          const float* __restrict__ A2,
          const float* __restrict__ W1, const float* __restrict__ W2,
          const float* __restrict__ bias,
          float* __restrict__ outp, int ldc, int cout,
          float* __restrict__ stats) {
  __shared__ __align__(16) _Float16 sW1[64 * LDA];
  __shared__ __align__(16) _Float16 sW2[64 * LDA];
  __shared__ __align__(16) _Float16 sA1[16 * LDA];
  __shared__ __align__(16) _Float16 sA2[16 * LDA];

  const int tid  = threadIdx.x;
  const int nthr = blockDim.x;
  const int NT   = nthr >> 5;  // n-tiles (4 for cout 64, 3 for 40)
  const int m0   = blockIdx.x * 16;

  // stage weights (f32 -> f16), pad rows >= cout with zeros
  const int wtot = NT * 16 * CIN;
  for (int i = tid; i < wtot; i += nthr) {
    int r = i / CIN, c = i - r * CIN;
    _Float16 w1v = (_Float16)0.f, w2v = (_Float16)0.f;
    if (r < cout) {
      if (HAS1) w1v = (_Float16)W1[r * CIN + c];
      w2v = (_Float16)W2[r * CIN + c];
    }
    if (HAS1) sW1[r * LDA + c] = w1v;
    sW2[r * LDA + c] = w2v;
  }
  // stage 16-row activation tiles (scale agg by 1/max(deg,1))
  for (int i = tid; i < 16 * CIN; i += nthr) {
    int r = i / CIN, c = i - r * CIN;
    int m = m0 + r;
    float a1v = 0.f, a2v = 0.f;
    if (m < NN) {
      if (HAS1) a1v = A1[(size_t)m * CIN + c] / fmaxf(deg[m], 1.0f);
      a2v = A2[(size_t)m * CIN + c];
    }
    if (HAS1) sA1[r * LDA + c] = (_Float16)a1v;
    sA2[r * LDA + c] = (_Float16)a2v;
  }
  __syncthreads();

  const int wave = tid >> 5;
  const int lane = tid & 31;
  const int lm   = lane & 15;
  const int hi   = lane >> 4;
  const int n0   = wave * 16;

  v8f acc = {};
  // ISA 16-bit A 16x32 layout: lane row lm; halves 0..7 -> k = hi*8+0..7,
  // halves 8..15 -> k = hi*8+16..23. B 32x16: lane col lm; same k order as A.
  const _Float16* a1b = sA1 + lm * LDA + hi * 8;
  const _Float16* a2b = sA2 + lm * LDA + hi * 8;
  const _Float16* w1b = sW1 + (n0 + lm) * LDA + hi * 8;
  const _Float16* w2b = sW2 + (n0 + lm) * LDA + hi * 8;

#pragma unroll
  for (int ks = 0; ks < CIN; ks += 32) {
    v16h bf2 = cat16(*(const v8h*)(w2b + ks), *(const v8h*)(w2b + ks + 16));
    v16h af2 = cat16(*(const v8h*)(a2b + ks), *(const v8h*)(a2b + ks + 16));
    acc = __builtin_amdgcn_wmma_f32_16x16x32_f16(false, af2, false, bf2,
                                                 (short)0, acc, false, false);
    if (HAS1) {
      v16h bf1 = cat16(*(const v8h*)(w1b + ks), *(const v8h*)(w1b + ks + 16));
      v16h af1 = cat16(*(const v8h*)(a1b + ks), *(const v8h*)(a1b + ks + 16));
      acc = __builtin_amdgcn_wmma_f32_16x16x32_f16(false, af1, false, bf1,
                                                   (short)0, acc, false, false);
    }
  }

  // epilogue: bias, store, optional BN sum/sumsq
  const int n = n0 + lm;
  float bval = (n < cout) ? bias[n] : 0.f;
  float s = 0.f, sq = 0.f;
#pragma unroll
  for (int r = 0; r < 8; ++r) {
    int m = m0 + hi * 8 + r;
    float v = acc[r] + bval;
    if (m < NN && n < cout) {
      outp[(size_t)m * ldc + n] = v;
      s += v;
      sq += v * v;
    }
  }
  if (stats) {
    s  += __shfl_xor(s, 16, 32);  // lanes l and l^16 hold the same column
    sq += __shfl_xor(sq, 16, 32);
    if (hi == 0 && n < cout) {
      atomicAdd(&stats[n], s);
      atomicAdd(&stats[64 + n], sq);
    }
  }
}

// ---------------------------------------------------------------------------
// BN: turn (sum, sumsq) into (scale, shift)
__global__ void bn_finalize(float* __restrict__ stats, const float* __restrict__ g,
                            const float* __restrict__ be) {
  int c = threadIdx.x;
  if (c < 64) {
    float mu  = stats[c] * (1.0f / NN);
    float var = stats[64 + c] * (1.0f / NN) - mu * mu;
    float sc  = g[c] * rsqrtf(var + 1e-5f);
    stats[128 + c] = sc;
    stats[192 + c] = be[c] - mu * sc;
  }
}

// BN apply + ReLU over N x 64 (row stride ld), float4 vectorized
__global__ void bn_apply(float* __restrict__ h, const float* __restrict__ stats, int ld) {
  int i = blockIdx.x * blockDim.x + threadIdx.x;  // over N*16 float4 chunks
  if (i >= NN * 16) return;
  int m  = i >> 4;
  int c4 = (i & 15) * 4;
  float* p = h + (size_t)m * ld + c4;
  float4 v = *(float4*)p;
  const float4 sc = *(const float4*)(stats + 128 + c4);
  const float4 sh = *(const float4*)(stats + 192 + c4);
  v.x = fmaxf(v.x * sc.x + sh.x, 0.f);
  v.y = fmaxf(v.y * sc.y + sh.y, 0.f);
  v.z = fmaxf(v.z * sc.z + sh.z, 0.f);
  v.w = fmaxf(v.w * sc.w + sh.w, 0.f);
  *(float4*)p = v;
}

// ---------------------------------------------------------------------------
// In-place log_softmax over 40 logits; one wave per row
__global__ void logsoftmax_kernel(float* __restrict__ outp) {
  int gid  = blockIdx.x * blockDim.x + threadIdx.x;
  int m    = gid >> 5;
  int lane = gid & 31;
  if (m >= NN) return;
  float v0 = (lane < 40)      ? outp[(size_t)m * 40 + lane]      : -3.4e38f;
  float v1 = (lane + 32 < 40) ? outp[(size_t)m * 40 + lane + 32] : -3.4e38f;
  float mx = fmaxf(v0, v1);
#pragma unroll
  for (int o = 16; o > 0; o >>= 1) mx = fmaxf(mx, __shfl_xor(mx, o, 32));
  float e0 = (lane < 40)      ? __expf(v0 - mx) : 0.f;
  float e1 = (lane + 32 < 40) ? __expf(v1 - mx) : 0.f;
  float sum = e0 + e1;
#pragma unroll
  for (int o = 16; o > 0; o >>= 1) sum += __shfl_xor(sum, o, 32);
  float ls = __logf(sum) + mx;
  if (lane < 40)      outp[(size_t)m * 40 + lane]      = v0 - ls;
  if (lane + 32 < 40) outp[(size_t)m * 40 + lane + 32] = v1 - ls;
}

// ---------------------------------------------------------------------------
extern "C" void kernel_launch(void* const* d_in, const int* in_sizes, int n_in,
                              void* d_out, int out_size, void* d_ws, size_t ws_size,
                              hipStream_t stream) {
  const float* x   = (const float*)d_in[0];
  const int*   esrc= (const int*)d_in[1];
  const int*   edst= (const int*)d_in[2];
  const float* af  = (const float*)d_in[3];
  const float* Wl0 = (const float*)d_in[4];
  const float* Wr0 = (const float*)d_in[5];
  const float* b0  = (const float*)d_in[6];
  const float* g0  = (const float*)d_in[7];
  const float* be0 = (const float*)d_in[8];
  const float* Wl1 = (const float*)d_in[9];
  const float* Wr1 = (const float*)d_in[10];
  const float* b1  = (const float*)d_in[11];
  const float* g1  = (const float*)d_in[12];
  const float* be1 = (const float*)d_in[13];
  const float* Wf  = (const float*)d_in[14];
  const float* bfb = (const float*)d_in[15];
  const float* Wl2 = (const float*)d_in[16];
  const float* Wr2 = (const float*)d_in[17];
  const float* b2  = (const float*)d_in[18];
  float* outp = (float*)d_out;

  // workspace layout (floats): deg | agg[N*128] | h0[N*64] | hcat[N*128] | stats
  float* ws   = (float*)d_ws;
  float* deg  = ws;
  float* agg  = deg + NN;
  float* h0   = agg + (size_t)NN * 128;
  float* hcat = h0 + (size_t)NN * 64;
  float* st0  = hcat + (size_t)NN * 128;
  float* st1  = st0 + 256;

  const int scat_blocks = (EE / EDGES_PER_WAVE) * 32 / 256;  // 4 edges per wave
  const int gemm_blocks = NN / 16;                           // 6250, exact

  hipMemsetAsync(deg, 0, NN * sizeof(float), stream);
  hipMemsetAsync(st0, 0, 512 * sizeof(float), stream);
  degree_kernel<<<(EE + 255) / 256, 256, 0, stream>>>(edst, deg);

  // ---- layer 0: SAGE(x) -> h0, BN+ReLU
  hipMemsetAsync(agg, 0, (size_t)NN * 128 * sizeof(float), stream);
  scatter_kernel<<<scat_blocks, 256, 0, stream>>>(x, esrc, edst, agg, 128);
  sage_gemm<128, true><<<gemm_blocks, 128, 0, stream>>>(agg, deg, x, Wl0, Wr0, b0,
                                                        h0, 64, 64, st0);
  bn_finalize<<<1, 64, 0, stream>>>(st0, g0, be0);
  bn_apply<<<(NN * 16 + 255) / 256, 256, 0, stream>>>(h0, st0, 64);

  // ---- layer 1: SAGE(h0) -> hcat[:, :64], BN+ReLU
  hipMemsetAsync(agg, 0, (size_t)NN * 64 * sizeof(float), stream);
  scatter_kernel<<<scat_blocks, 256, 0, stream>>>(h0, esrc, edst, agg, 64);
  sage_gemm<64, true><<<gemm_blocks, 128, 0, stream>>>(agg, deg, h0, Wl1, Wr1, b1,
                                                       hcat, 128, 64, st1);
  bn_finalize<<<1, 64, 0, stream>>>(st1, g1, be1);
  bn_apply<<<(NN * 16 + 255) / 256, 256, 0, stream>>>(hcat, st1, 128);

  // ---- dense projection: hcat[:, 64:] = add_feat @ Wf^T + bf
  sage_gemm<128, false><<<gemm_blocks, 128, 0, stream>>>(nullptr, nullptr, af,
                                                         nullptr, Wf, bfb,
                                                         hcat + 64, 128, 64, nullptr);

  // ---- layer 2: SAGE(hcat) -> logits (d_out), then log_softmax in place
  hipMemsetAsync(agg, 0, (size_t)NN * 128 * sizeof(float), stream);
  scatter_kernel<<<scat_blocks, 256, 0, stream>>>(hcat, esrc, edst, agg, 128);
  sage_gemm<128, true><<<gemm_blocks, 96, 0, stream>>>(agg, deg, hcat, Wl2, Wr2, b2,
                                                       outp, 40, 40, nullptr);
  logsoftmax_kernel<<<(NN * 32 + 255) / 256, 256, 0, stream>>>(outp);
}
